// MutiScaleDictionaryCrossAttentionGLU_35304631173891
// MI455X (gfx1250) — matrix operand order repaired and checked by reading, compiled, not run on hardware
//
#include <hip/hip_runtime.h>

// ---------------------------------------------------------------------------
// Types (trivial ext-vectors so they can live in unions)
// ---------------------------------------------------------------------------
typedef __attribute__((ext_vector_type(16))) __bf16 v16bf;
typedef __attribute__((ext_vector_type(8)))  float  v8f;
typedef unsigned int u32x4 __attribute__((ext_vector_type(4)));
typedef int          v4i   __attribute__((ext_vector_type(4)));

#define PTOT  32768      // 8 * 64 * 64 pixels
#define DDIM  640
#define HIDD  1280
#define NE    20         // heads
#define CCH   32         // head dim
#define NTOK  64         // dict tokens per batch
#define HWSZ  4096       // 64*64
#define BATCH 8

#ifndef __has_builtin
#define __has_builtin(x) 0
#endif
#if __has_builtin(__builtin_amdgcn_global_load_async_to_lds_b128) && \
    __has_builtin(__builtin_amdgcn_s_wait_asynccnt)
#define ASYNC_LDS 1
#else
#define ASYNC_LDS 0
#endif

__device__ __forceinline__ unsigned short f2bf(float f) {
  unsigned int u = __float_as_uint(f);
  unsigned int r = u + 0x7FFFu + ((u >> 16) & 1u);   // round-to-nearest-even
  return (unsigned short)(r >> 16);
}
__device__ __forceinline__ float bf2f(unsigned short s) {
  return __uint_as_float((unsigned int)s << 16);
}
__device__ __forceinline__ float geluf(float x) {
  return 0.5f * x * (1.0f + erff(x * 0.70710678118654752f));
}

#if ASYNC_LDS
__device__ __forceinline__ void async_b128(void* lds, const void* g) {
  __builtin_amdgcn_global_load_async_to_lds_b128(
      (__attribute__((address_space(1))) v4i*)g,
      (__attribute__((address_space(3))) v4i*)lds, 0, 0);
}
#endif

union FragBF {
  unsigned short s[16];
  u32x4 q[2];
  v16bf v;
};

// ---------------------------------------------------------------------------
// Tiled bf16 WMMA GEMM:  C(MxN) = epilogue( A(MxK,bf16) @ Bw(packed bf16) )
// Block 256 thr (8 waves). WG tile 256x64, wave tile 64x32 (4x2 wmma), BK=32.
// Double-buffered LDS fed by GLOBAL_LOAD_ASYNC_TO_LDS_B128: tile k+1 streams
// while tile k computes; s_wait_asynccnt(5) gates on the *previous* tile only.
// Bw is tile-packed: [n0/64][k0/32][n:64][k:32] (4KB contiguous per tile).
// MODE 0: C = acc + bias
// MODE 1: C = gelu(acc + bias)
// MODE 2: C = acc + bias + res[m,n]*rs[n]     (res is f32)
// MODE 3: C += acc  (f32 accumulate, no bias)
// MODE 4: NCHW f32 store: C[((b*N+n)*4096)+hw] = acc + bias
// OBF: 1 -> store bf16, 0 -> store f32  (MODE 3/4 always f32)
// M%256==0, N%64==0, K%32==0 for every call here.
// ---------------------------------------------------------------------------
template<int MODE, int OBF>
__global__ __launch_bounds__(256) void gemm_bf16_k(
    const unsigned short* __restrict__ A, const unsigned short* __restrict__ Bw,
    const float* __restrict__ bias, void* __restrict__ Cv,
    int M, int N, int K,
    const float* __restrict__ res, const float* __restrict__ rs)
{
  __shared__ alignas(16) unsigned short sA[2][256 * 32];   // [m][k]
  __shared__ alignas(16) unsigned short sB[2][64 * 32];    // [n][k]

  const int tid  = threadIdx.x;
  const int lane = tid & 31;
  const int wid  = tid >> 5;
  const int m0   = blockIdx.x * 256;
  const int n0   = blockIdx.y * 64;
  const int wm   = (wid & 3) * 64;      // wave M offset (4-way split of 256)
  const int wn   = (wid >> 2) * 32;     // wave N offset (2-way split of 64)
  const int KT   = K >> 5;

  v8f acc[4][2] = {};

  const int lm     = lane & 15;
  const int khalfA = (lane < 16) ? 0 : 8;    // A frag K-half (ISA 16-bit A layout)
  const int kbaseB = (lane < 16) ? 0 : 8;    // B frag K-half (same layout as A)

  const unsigned short* agBase = A + (size_t)(m0 + tid) * K;
  const size_t bgBase = ((size_t)(n0 >> 6) * KT) << 11;

  auto issue = [&](int kt, int buf) {
    const unsigned short* ag = agBase + (kt << 5);         // 64B row chunk
    unsigned short*       al = &sA[buf][tid * 32];
    const unsigned short* bg = Bw + bgBase + ((size_t)kt << 11) + tid * 8;
    unsigned short*       bl = &sB[buf][tid * 8];
#if ASYNC_LDS
    async_b128(al,      ag);
    async_b128(al + 8,  ag + 8);
    async_b128(al + 16, ag + 16);
    async_b128(al + 24, ag + 24);
    async_b128(bl, bg);
#else
    *(u32x4*)(al)      = *(const u32x4*)(ag);
    *(u32x4*)(al + 8)  = *(const u32x4*)(ag + 8);
    *(u32x4*)(al + 16) = *(const u32x4*)(ag + 16);
    *(u32x4*)(al + 24) = *(const u32x4*)(ag + 24);
    *(u32x4*)(bl)      = *(const u32x4*)(bg);
#endif
  };

  issue(0, 0);
  for (int kt = 0; kt < KT; ++kt) {
    const int cur = kt & 1;
    if (kt + 1 < KT) {
      issue(kt + 1, cur ^ 1);
#if ASYNC_LDS
      __builtin_amdgcn_s_wait_asynccnt(5);   // only tile kt's 5 loads must land
#endif
    } else {
#if ASYNC_LDS
      __builtin_amdgcn_s_wait_asynccnt(0);
#endif
    }
    __syncthreads();

    FragBF a[4], b[2];
#pragma unroll
    for (int i = 0; i < 4; ++i) {
      const unsigned short* ap = &sA[cur][(wm + i * 16 + lm) * 32];
      a[i].q[0] = *(const u32x4*)(ap + khalfA);
      a[i].q[1] = *(const u32x4*)(ap + 16 + khalfA);
    }
#pragma unroll
    for (int j = 0; j < 2; ++j) {
      const unsigned short* bp = &sB[cur][(wn + j * 16 + lm) * 32 + kbaseB];
      b[j].q[0] = *(const u32x4*)(bp);
      b[j].q[1] = *(const u32x4*)(bp + 16);
    }
#pragma unroll
    for (int i = 0; i < 4; ++i)
#pragma unroll
      for (int j = 0; j < 2; ++j)
        acc[i][j] = __builtin_amdgcn_wmma_f32_16x16x32_bf16(
            false, a[i].v, false, b[j].v, (short)0, acc[i][j], false, false);
    __syncthreads();   // everyone done reading buf[cur] before it is refilled
  }

  float* Cf = (float*)Cv;
  unsigned short* Cb = (unsigned short*)Cv;
  const int mh = (lane < 16) ? 0 : 8;
#pragma unroll
  for (int i = 0; i < 4; ++i) {
#pragma unroll
    for (int j = 0; j < 2; ++j) {
      const int nn = n0 + wn + j * 16 + lm;
#pragma unroll
      for (int r = 0; r < 8; ++r) {
        const int mm = m0 + wm + i * 16 + mh + r;
        const size_t idx = (size_t)mm * N + nn;
        float v = acc[i][j][r];
        if (MODE == 3) {
          Cf[idx] += v;
        } else {
          v += bias[nn];
          if (MODE == 1) v = geluf(v);
          if (MODE == 2) v += res[idx] * rs[nn];
          if (MODE == 4) {
            const int bb = mm >> 12, hw = mm & 4095;
            Cf[((size_t)bb * N + nn) * HWSZ + hw] = v;
          } else if (OBF) {
            Cb[idx] = f2bf(v);
          } else {
            Cf[idx] = v;
          }
        }
      }
    }
  }
}

// ---------------------------------------------------------------------------
// Cross-attention, WMMA path (all operands bf16 in global).
// Per block: one (b,e) pair, 128 query rows (8 waves x 16 rows).
// ---------------------------------------------------------------------------
__global__ __launch_bounds__(256) void attn_k(
    const unsigned short* __restrict__ Q,   // (P, 640) bf16
    const unsigned short* __restrict__ Kb,  // (512, 640) bf16
    const unsigned short* __restrict__ Vb,  // (512, 640) bf16 (dtn)
    const float* __restrict__ scale,
    unsigned short* __restrict__ O)         // (P, 640) bf16
{
  __shared__ alignas(16) unsigned short pl[8 * 16 * 64];  // per-wave P tile (bf16)

  const int tid = threadIdx.x, lane = tid & 31, wid = tid >> 5;
  const int bid   = blockIdx.x;
  const int chunk = bid & 31;
  const int be    = bid >> 5;
  const int b     = be / NE, e = be % NE;
  const int mbase = chunk * 128 + wid * 16;

  const int lm    = lane & 15;
  const int khalf = (lane < 16) ? 0 : 8;
  const int kbase = (lane < 16) ? 0 : 8;
  const int mh    = (lane < 16) ? 0 : 8;

  FragBF qa;
  {
    const unsigned short* qp = Q + ((size_t)b * HWSZ + mbase + lm) * DDIM + e * CCH;
    qa.q[0] = *(const u32x4*)(qp + khalf);
    qa.q[1] = *(const u32x4*)(qp + 16 + khalf);
  }
  FragBF kf[4];
#pragma unroll
  for (int t = 0; t < 4; ++t) {
    const unsigned short* kp = Kb + ((size_t)b * NTOK + t * 16 + lm) * DDIM + e * CCH + kbase;
    kf[t].q[0] = *(const u32x4*)(kp);
    kf[t].q[1] = *(const u32x4*)(kp + 16);
  }

  v8f sim[4];
  {
    v8f z = {};
#pragma unroll
    for (int t = 0; t < 4; ++t)
      sim[t] = __builtin_amdgcn_wmma_f32_16x16x32_bf16(
          false, qa.v, false, kf[t].v, (short)0, z, false, false);
  }

  const float sc = scale[e];
  unsigned short* pw = pl + wid * 1024;

#pragma unroll
  for (int r = 0; r < 8; ++r) {
    float mx = -1e30f;
#pragma unroll
    for (int t = 0; t < 4; ++t) { sim[t][r] *= sc; mx = fmaxf(mx, sim[t][r]); }
#pragma unroll
    for (int d = 1; d < 16; d <<= 1) mx = fmaxf(mx, __shfl_xor(mx, d, 32));
    float sum = 0.0f;
#pragma unroll
    for (int t = 0; t < 4; ++t) { float ev = __expf(sim[t][r] - mx); sim[t][r] = ev; sum += ev; }
#pragma unroll
    for (int d = 1; d < 16; d <<= 1) sum += __shfl_xor(sum, d, 32);
    const float inv = 1.0f / sum;
#pragma unroll
    for (int t = 0; t < 4; ++t)
      pw[(mh + r) * 64 + t * 16 + lm] = f2bf(sim[t][r] * inv);
  }
  __syncthreads();

  v8f oacc[2] = {};
#pragma unroll
  for (int ks = 0; ks < 64; ks += 32) {
    FragBF pa;
    const unsigned short* pp = pw + lm * 64 + ks;
    pa.q[0] = *(const u32x4*)(pp + khalf);
    pa.q[1] = *(const u32x4*)(pp + 16 + khalf);
#pragma unroll
    for (int ct = 0; ct < 2; ++ct) {
      FragBF vf;
#pragma unroll
      for (int j = 0; j < 16; ++j)
        vf.s[j] = Vb[((size_t)b * NTOK + ks + kbase + j + (j & 8)) * DDIM + e * CCH + ct * 16 + lm];
      oacc[ct] = __builtin_amdgcn_wmma_f32_16x16x32_bf16(
          false, pa.v, false, vf.v, (short)0, oacc[ct], false, false);
    }
  }
#pragma unroll
  for (int ct = 0; ct < 2; ++ct)
#pragma unroll
    for (int r = 0; r < 8; ++r)
      O[((size_t)b * HWSZ + mbase + mh + r) * DDIM + e * CCH + ct * 16 + lm] =
          f2bf(oacc[ct][r]);
}

// ---------------------------------------------------------------------------
// LayerNorm: one wave per row; f32 in, bf16 out (always feeds a GEMM).
// ---------------------------------------------------------------------------
__global__ __launch_bounds__(256) void ln_k(
    const float* __restrict__ x, const float* __restrict__ g,
    const float* __restrict__ b, unsigned short* __restrict__ y, int C, int rows)
{
  const int lane = threadIdx.x & 31;
  const int row  = blockIdx.x * 8 + (threadIdx.x >> 5);
  if (row >= rows) return;
  const float* xr = x + (size_t)row * C;
  float s = 0.0f, s2 = 0.0f;
  for (int j = lane; j < C; j += 32) { float v = xr[j]; s += v; s2 += v * v; }
#pragma unroll
  for (int d = 1; d < 32; d <<= 1) { s += __shfl_xor(s, d, 32); s2 += __shfl_xor(s2, d, 32); }
  const float mean = s / (float)C;
  const float var  = s2 / (float)C - mean * mean;
  const float inv  = rsqrtf(var + 1e-5f);
  unsigned short* yr = y + (size_t)row * C;
  for (int j = lane; j < C; j += 32)
    yr[j] = f2bf((xr[j] - mean) * inv * g[j] + b[j]);
}

// NCHW (B,Cin,64,64) f32 -> (pixel, channel) bf16
__global__ void nchw2pc_k(const float* __restrict__ x, unsigned short* __restrict__ xT, int Cin)
{
  const int idx = blockIdx.x * 256 + threadIdx.x;
  if (idx >= BATCH * Cin * HWSZ) return;
  const int b  = idx / (Cin * HWSZ);
  const int r  = idx - b * Cin * HWSZ;
  const int c  = r / HWSZ;
  const int hw = r - c * HWSZ;
  xT[((size_t)b * HWSZ + hw) * Cin + c] = f2bf(x[idx]);
}

// (K,N) f32 (row stride ss) -> tile-packed bf16 [n/64][k/32][n:64][k:32]
__global__ void pack_w_k(const float* __restrict__ src, unsigned short* __restrict__ dst,
                         int K, int N, int ss)
{
  const int idx = blockIdx.x * 256 + threadIdx.x;
  if (idx >= K * N) return;
  const int k = idx / N, n = idx - k * N;
  dst[(((size_t)(n >> 6) * (K >> 5) + (k >> 5)) << 11) + ((n & 63) << 5) + (k & 31)] =
      f2bf(src[(size_t)k * ss + n]);
}
// (N,K) f32 (row stride ss; conv OIHW 1x1 weights) -> same packed layout
__global__ void pack_wT_k(const float* __restrict__ src, unsigned short* __restrict__ dst,
                          int K, int N, int ss)
{
  const int idx = blockIdx.x * 256 + threadIdx.x;
  if (idx >= K * N) return;
  const int n = idx / K, k = idx - n * K;
  dst[(((size_t)(n >> 6) * (K >> 5) + (k >> 5)) << 11) + ((n & 63) << 5) + (k & 31)] =
      f2bf(src[(size_t)n * ss + k]);
}

// bf16 gelu elementwise
__global__ void gelu_k(const unsigned short* __restrict__ x, unsigned short* __restrict__ y, int n)
{
  const int i = blockIdx.x * 256 + threadIdx.x;
  if (i < n) y[i] = f2bf(geluf(bf2f(x[i])));
}

// Depthwise 3x3 (pad 1), bf16 in/out, gelu fused; optional GLU gate.
template<bool GLU>
__global__ void dw3x3_k(const unsigned short* __restrict__ src, int srcStride,
                        const float* __restrict__ w, const float* __restrict__ bias,
                        const unsigned short* __restrict__ gate, int gateOff,
                        unsigned short* __restrict__ out, int outStride, int C)
{
  const int idx = blockIdx.x * 256 + threadIdx.x;
  if (idx >= PTOT * C) return;
  const int p = idx / C, c = idx - p * C;
  const int h = (p >> 6) & 63, ww = p & 63;
  float s = bias[c];
#pragma unroll
  for (int dy = -1; dy <= 1; ++dy) {
#pragma unroll
    for (int dx = -1; dx <= 1; ++dx) {
      if ((unsigned)(h + dy) < 64u && (unsigned)(ww + dx) < 64u)
        s += bf2f(src[(size_t)(p + dy * 64 + dx) * srcStride + c]) *
             w[c * 9 + (dy + 1) * 3 + (dx + 1)];
    }
  }
  float g = geluf(s);
  if (GLU) g *= bf2f(gate[(size_t)p * srcStride + gateOff + c]);
  out[(size_t)p * outStride + c] = f2bf(g);
}

// channel mean & max per pixel (one wave per pixel), f32 s_out
__global__ __launch_bounds__(256) void avgmax_k(
    const float* __restrict__ s, float* __restrict__ avg, float* __restrict__ mx)
{
  const int lane = threadIdx.x & 31;
  const int p = blockIdx.x * 8 + (threadIdx.x >> 5);
  if (p >= PTOT) return;
  const float* sr = s + (size_t)p * DDIM;
  float a = 0.0f, m = -1e30f;
  for (int j = lane; j < DDIM; j += 32) { float v = sr[j]; a += v; m = fmaxf(m, v); }
#pragma unroll
  for (int d = 1; d < 32; d <<= 1) { a += __shfl_xor(a, d, 32); m = fmaxf(m, __shfl_xor(m, d, 32)); }
  if (lane == 0) { avg[p] = a / (float)DDIM; mx[p] = m; }
}

// 7x7 conv over {avg,max} (pad 3) + sigmoid
__global__ void saconv_k(const float* __restrict__ avg, const float* __restrict__ mx,
                         const float* __restrict__ w, float* __restrict__ att)
{
  const int p = blockIdx.x * 256 + threadIdx.x;
  if (p >= PTOT) return;
  const int b = p >> 12, hw = p & 4095, h = hw >> 6, ww = hw & 63;
  float s = 0.0f;
  for (int dy = -3; dy <= 3; ++dy) {
    for (int dx = -3; dx <= 3; ++dx) {
      if ((unsigned)(h + dy) < 64u && (unsigned)(ww + dx) < 64u) {
        const int q = b * HWSZ + (h + dy) * 64 + (ww + dx);
        const int wi = (dy + 3) * 7 + (dx + 3);
        s += avg[q] * w[wi] + mx[q] * w[49 + wi];
      }
    }
  }
  att[p] = 1.0f / (1.0f + __expf(-s));
}

// x1 = s_out * att + x * rs1   (in-place into x)
__global__ void msaout_k(const float* __restrict__ sout, const float* __restrict__ att,
                         float* __restrict__ x, const float* __restrict__ rs1)
{
  const int idx = blockIdx.x * 256 + threadIdx.x;
  if (idx >= PTOT * DDIM) return;
  const int p = idx / DDIM, c = idx - p * DDIM;
  x[idx] = sout[idx] * att[p] + x[idx] * rs1[c];
}

// ---------------------------------------------------------------------------
// Host launcher
// ---------------------------------------------------------------------------
extern "C" void kernel_launch(void* const* d_in, const int* in_sizes, int n_in,
                              void* d_out, int out_size, void* d_ws, size_t ws_size,
                              hipStream_t stream) {
  (void)in_sizes; (void)n_in; (void)out_size; (void)ws_size;
  const float* x_in     = (const float*)d_in[0];
  const float* dt_in    = (const float*)d_in[1];
  const float* p_scale  = (const float*)d_in[2];
  const float* xtrans_w = (const float*)d_in[3];
  const float* xtrans_b = (const float*)d_in[4];
  const float* lnsc_g   = (const float*)d_in[5];
  const float* lnsc_b   = (const float*)d_in[6];
  const float* s_w      = (const float*)d_in[7];
  const float* s_b      = (const float*)d_in[8];
  const float* sa_w     = (const float*)d_in[9];
  const float* din_w    = (const float*)d_in[10];
  const float* din_b    = (const float*)d_in[11];
  const float* ddw_w    = (const float*)d_in[12];
  const float* ddw_b    = (const float*)d_in[13];
  const float* dout_w   = (const float*)d_in[14];
  const float* dout_b   = (const float*)d_in[15];
  const float* proj_w   = (const float*)d_in[16];
  const float* proj_b   = (const float*)d_in[17];
  const float* lnx_g    = (const float*)d_in[18];
  const float* lnx_b    = (const float*)d_in[19];
  const float* q_w      = (const float*)d_in[20];
  const float* q_b      = (const float*)d_in[21];
  const float* dln_g    = (const float*)d_in[22];
  const float* dln_b    = (const float*)d_in[23];
  const float* k_w      = (const float*)d_in[24];
  const float* k_b      = (const float*)d_in[25];
  const float* lin_w    = (const float*)d_in[26];
  const float* lin_b    = (const float*)d_in[27];
  const float* lnm_g    = (const float*)d_in[28];
  const float* lnm_b    = (const float*)d_in[29];
  const float* fc1_w    = (const float*)d_in[30];
  const float* fc1_b    = (const float*)d_in[31];
  const float* mdw_w    = (const float*)d_in[32];
  const float* mdw_b    = (const float*)d_in[33];
  const float* fc2_w    = (const float*)d_in[34];
  const float* fc2_b    = (const float*)d_in[35];
  const float* out_w    = (const float*)d_in[36];
  const float* out_b    = (const float*)d_in[37];
  const float* rs1      = (const float*)d_in[38];
  const float* rs2      = (const float*)d_in[39];
  const float* rs3      = (const float*)d_in[40];

  // ---- workspace carve-up --------------------------------------------------
  size_t off = 0;
  char* basep = (char*)d_ws;
  auto alloc = [&](size_t bytes) -> void* {
    void* p = basep + off; off += (bytes + 255) & ~(size_t)255; return p;
  };
  float* bufX = (float*)alloc((size_t)PTOT * DDIM * 4);   // x / x1 residual trunk
  float* sOut = (float*)alloc((size_t)PTOT * DDIM * 4);   // msa s_out, then out2
  float* avgb = (float*)alloc((size_t)PTOT * 4);
  float* mxb  = (float*)alloc((size_t)PTOT * 4);
  float* attb = (float*)alloc((size_t)PTOT * 4);
  unsigned short* xTb = (unsigned short*)alloc((size_t)PTOT * 320 * 2);
  unsigned short* bA  = (unsigned short*)alloc((size_t)PTOT * DDIM * 2);
  unsigned short* bB  = (unsigned short*)alloc((size_t)PTOT * DDIM * 2);
  unsigned short* bP  = (unsigned short*)alloc((size_t)PTOT * DDIM * 2);
  unsigned short* dtn = (unsigned short*)alloc((size_t)512 * DDIM * 2);
  unsigned short* kb  = (unsigned short*)alloc((size_t)512 * DDIM * 2);
  unsigned short* Hb  = (unsigned short*)alloc((size_t)PTOT * 2560 * 2);
  unsigned short* Gb  = (unsigned short*)alloc((size_t)PTOT * HIDD * 2);
  unsigned short* wXT   = (unsigned short*)alloc((size_t)320 * 640 * 2);
  unsigned short* wS    = (unsigned short*)alloc((size_t)640 * 640 * 2);
  unsigned short* wDIn  = (unsigned short*)alloc((size_t)3 * 640 * 640 * 2);
  unsigned short* wDOut = (unsigned short*)alloc((size_t)3 * 640 * 640 * 2);
  unsigned short* wProj = (unsigned short*)alloc((size_t)4 * 640 * 640 * 2);
  unsigned short* wQ    = (unsigned short*)alloc((size_t)640 * 640 * 2);
  unsigned short* wK    = (unsigned short*)alloc((size_t)640 * 640 * 2);
  unsigned short* wLin  = (unsigned short*)alloc((size_t)640 * 640 * 2);
  unsigned short* wFc1  = (unsigned short*)alloc((size_t)640 * 2560 * 2);
  unsigned short* wFc2  = (unsigned short*)alloc((size_t)1280 * 640 * 2);
  unsigned short* wOut  = (unsigned short*)alloc((size_t)640 * 320 * 2);

  auto blks = [](int n) { return (n + 255) / 256; };
  const size_t W64 = (size_t)640 * 640;

  // ---- weight packing to tile-major bf16 -----------------------------------
  pack_w_k<<<blks(320 * 640), 256, 0, stream>>>(xtrans_w, wXT, 320, 640, 640);
  pack_wT_k<<<blks(640 * 640), 256, 0, stream>>>(s_w, wS, 640, 640, 640);
  for (int i = 0; i < 3; ++i) {
    pack_wT_k<<<blks(640 * 640), 256, 0, stream>>>(din_w + i * W64, wDIn + i * W64, 640, 640, 640);
    pack_wT_k<<<blks(640 * 640), 256, 0, stream>>>(dout_w + i * W64, wDOut + i * W64, 640, 640, 640);
  }
  for (int i = 0; i < 4; ++i)  // proj chunks: input channels i*640..(i+1)*640
    pack_wT_k<<<blks(640 * 640), 256, 0, stream>>>(proj_w + i * 640, wProj + i * W64, 640, 640, 2560);
  pack_w_k<<<blks(640 * 640), 256, 0, stream>>>(q_w, wQ, 640, 640, 640);
  pack_w_k<<<blks(640 * 640), 256, 0, stream>>>(k_w, wK, 640, 640, 640);
  pack_w_k<<<blks(640 * 640), 256, 0, stream>>>(lin_w, wLin, 640, 640, 640);
  pack_w_k<<<blks(640 * 2560), 256, 0, stream>>>(fc1_w, wFc1, 640, 2560, 2560);
  pack_w_k<<<blks(1280 * 640), 256, 0, stream>>>(fc2_w, wFc2, 1280, 640, 640);
  pack_w_k<<<blks(640 * 320), 256, 0, stream>>>(out_w, wOut, 640, 320, 320);

  const dim3 g640(PTOT / 256, 640 / 64);
  const int nPD = PTOT * DDIM;

  // ---- 1) x transpose + x_trans GEMM --------------------------------------
  nchw2pc_k<<<blks(BATCH * 320 * HWSZ), 256, 0, stream>>>(x_in, xTb, 320);
  gemm_bf16_k<0, 0><<<g640, 256, 0, stream>>>(xTb, wXT, xtrans_b, bufX,
                                              PTOT, 640, 320, nullptr, nullptr);

  // ---- 2) MSA --------------------------------------------------------------
  ln_k<<<PTOT / 8, 256, 0, stream>>>(bufX, lnsc_g, lnsc_b, bA, DDIM, PTOT);
  gemm_bf16_k<0, 1><<<g640, 256, 0, stream>>>(bA, wS, s_b, bP,
                                              PTOT, 640, 640, nullptr, nullptr);   // s
  gemm_bf16_k<0, 0><<<g640, 256, 0, stream>>>(bP, wProj, proj_b, sOut,
                                              PTOT, 640, 640, nullptr, nullptr);   // proj chunk 0
  for (int i = 0; i < 3; ++i) {
    gelu_k<<<blks(nPD), 256, 0, stream>>>(bP, bA, nPD);
    gemm_bf16_k<1, 1><<<g640, 256, 0, stream>>>(bA, wDIn + i * W64,
                                                din_b + (size_t)i * 640, bB,
                                                PTOT, 640, 640, nullptr, nullptr); // gelu(dense_in)
    dw3x3_k<false><<<blks(nPD), 256, 0, stream>>>(bB, 640,
                                                  ddw_w + (size_t)i * 640 * 9,
                                                  ddw_b + (size_t)i * 640,
                                                  nullptr, 0, bA, 640, 640);       // gelu(dwconv)
    gemm_bf16_k<0, 1><<<g640, 256, 0, stream>>>(bA, wDOut + i * W64,
                                                dout_b + (size_t)i * 640, bP,
                                                PTOT, 640, 640, nullptr, nullptr); // dense_out
    gemm_bf16_k<3, 0><<<g640, 256, 0, stream>>>(bP, wProj + (size_t)(i + 1) * W64,
                                                nullptr, sOut,
                                                PTOT, 640, 640, nullptr, nullptr); // proj accumulate
  }
  avgmax_k<<<PTOT / 8, 256, 0, stream>>>(sOut, avgb, mxb);
  saconv_k<<<blks(PTOT), 256, 0, stream>>>(avgb, mxb, sa_w, attb);
  msaout_k<<<blks(nPD), 256, 0, stream>>>(sOut, attb, bufX, rs1);                  // x1 (in place)

  // ---- 3) cross attention --------------------------------------------------
  ln_k<<<PTOT / 8, 256, 0, stream>>>(bufX, lnx_g, lnx_b, bA, DDIM, PTOT);
  gemm_bf16_k<0, 1><<<g640, 256, 0, stream>>>(bA, wQ, q_b, bP,
                                              PTOT, 640, 640, nullptr, nullptr);   // q
  ln_k<<<512 / 8, 256, 0, stream>>>(dt_in, dln_g, dln_b, dtn, DDIM, 512);
  gemm_bf16_k<0, 1><<<dim3(512 / 256, 640 / 64), 256, 0, stream>>>(
      dtn, wK, k_b, kb, 512, 640, 640, nullptr, nullptr);                          // k
  attn_k<<<BATCH * NE * 32, 256, 0, stream>>>(bP, kb, dtn, p_scale, bA);           // attn out
  gemm_bf16_k<2, 0><<<g640, 256, 0, stream>>>(bA, wLin, lin_b, sOut,
                                              PTOT, 640, 640, bufX, rs2);          // out2

  // ---- 4) GLU MLP ----------------------------------------------------------
  ln_k<<<PTOT / 8, 256, 0, stream>>>(sOut, lnm_g, lnm_b, bA, DDIM, PTOT);
  gemm_bf16_k<0, 1><<<dim3(PTOT / 256, 2560 / 64), 256, 0, stream>>>(
      bA, wFc1, fc1_b, Hb, PTOT, 2560, 640, nullptr, nullptr);                     // h
  dw3x3_k<true><<<blks(PTOT * HIDD), 256, 0, stream>>>(Hb, 2560, mdw_w, mdw_b,
                                                       Hb, HIDD, Gb, HIDD, HIDD);  // gelu(dw)*hv
  gemm_bf16_k<2, 1><<<g640, 256, 0, stream>>>(Gb, wFc2, fc2_b, bB,
                                              PTOT, 640, 1280, sOut, rs3);         // mlp + out2*rs3

  // ---- 5) output projection + NCHW store ----------------------------------
  gemm_bf16_k<4, 0><<<dim3(PTOT / 256, 320 / 64), 256, 0, stream>>>(
      bB, wOut, out_b, d_out, PTOT, 320, 640, nullptr, nullptr);
}
